// HGNND_31353261260882
// MI455X (gfx1250) — matrix operand order repaired and checked by reading, not compiled, not run on hardware
//
#include <hip/hip_runtime.h>
#include <hip/hip_bf16.h>
#include <math.h>

typedef __attribute__((ext_vector_type(16))) __bf16 v16bf;
typedef __attribute__((ext_vector_type(8)))  __bf16 v8bf;
typedef __attribute__((ext_vector_type(8)))  float  v8f;

#define HID 128
#define KCHUNK 128
#define BPAD 8
#define BSTRIDE (KCHUNK + BPAD)   // 136 bf16 = 272 B row stride (16B aligned, bank-staggered)

// ---------------------------------------------------------------------------
// One-time weight transform: Wt[n][k] = bf16(W[k][n]). Weights are tiny
// (<=128KB each as f32) so this is noise; it lets the GEMM stage B into LDS with
// pure async byte-moves (no f32->bf16 cvt in the hot loop, half the bytes).
// ---------------------------------------------------------------------------
__global__ void wt_transpose_kernel(const float* __restrict__ W, __bf16* __restrict__ Wt,
                                    int K, int NN) {
    int i = (int)(blockIdx.x * blockDim.x + threadIdx.x);
    if (i >= K * NN) return;
    int n = i / K, k = i - n * K;            // coalesced writes to Wt
    Wt[(size_t)n * K + k] = (__bf16)W[(size_t)k * NN + n];
}

// ---------------------------------------------------------------------------
// WMMA bf16 GEMM: C[M x NN] = A[M x K] * Bt^T (+ bias), f32 A/C, bf16 Bt.
// Bt is transposed bf16 [NN][K]. Block = 8 waves covering 128 M-rows x full NN.
// B chunks are DMA'd into LDS with GLOBAL_LOAD_ASYNC_TO_LDS_B128 (ASYNCcnt),
// then each lane's B fragment is two contiguous ds_load_b128. A fragments are
// loaded once per K-chunk and reused across all NN/16 tiles.
// v_wmma_f32_16x16x32_bf16 per ISA 7.12.2 layouts.
// ---------------------------------------------------------------------------
template<int K, int NN>
__global__ void wmma_gemm_kernel(const float* __restrict__ A,
                                 const __bf16* __restrict__ Bt,
                                 const float* __restrict__ bias,
                                 float* __restrict__ C, int M) {
    static_assert(K % KCHUNK == 0, "K multiple of 128");
    constexpr int tilesN = NN / 16;

    __shared__ __bf16 blds[NN * BSTRIDE];

    const int waveId  = threadIdx.x >> 5;            // 0..7
    const int lane    = threadIdx.x & 31;
    const int half    = lane >> 4;                   // 0: lanes 0-15, 1: lanes 16-31
    const int l15     = lane & 15;
    const int rowTile = blockIdx.x * 8 + waveId;     // 16-row tile owned by this wave
    const bool active = (rowTile * 16 < M);          // wave-uniform

    const float* arow = A + (size_t)(rowTile * 16 + l15) * K;

    v8f acc[tilesN];
#pragma unroll
    for (int tn = 0; tn < tilesN; ++tn) acc[tn] = {};

    for (int kc = 0; kc < K; kc += KCHUNK) {
        __syncthreads();   // protect blds reuse across chunks
        // --- async DMA B chunk: 16B chunks of transposed bf16 W -> LDS ---
        // chunk idx -> (n, koct): global elems n*K + kc + koct*8,
        //                         LDS elems    n*BSTRIDE + koct*8  (both 16B aligned)
        constexpr int CHUNKS = NN * (KCHUNK / 8);    // 16B transfers per K-chunk
        for (int idx = threadIdx.x; idx < CHUNKS; idx += 256) {
            int n    = idx >> 4;                     // KCHUNK/8 == 16
            int koct = idx & 15;
            unsigned lds_off = (unsigned)(unsigned long long)(const void*)
                               &blds[n * BSTRIDE + koct * 8];
            unsigned glb_off = (unsigned)(((size_t)n * K + kc + koct * 8) * sizeof(__bf16));
            asm volatile("global_load_async_to_lds_b128 %0, %1, %2"
                         :: "v"(lds_off), "v"(glb_off), "s"(Bt)
                         : "memory");
        }
        asm volatile("s_wait_asynccnt 0x0" ::: "memory");
        __syncthreads();

        if (active) {
            // --- A fragments for this chunk: 4 k-steps of 32 ---
            v16bf afr[4];
#pragma unroll
            for (int ks = 0; ks < 4; ++ks) {
#pragma unroll
                for (int g = 0; g < 2; ++g) {
                    const float* ap = arow + kc + ks * 32 + half * 16 + g * 8;
#pragma unroll
                    for (int j = 0; j < 8; ++j)
                        afr[ks][g * 8 + j] = (__bf16)ap[j];
                }
            }
            // --- sweep all N tiles, reusing A fragments ---
#pragma unroll
            for (int tn = 0; tn < tilesN; ++tn) {
                const __bf16* bbase = &blds[(tn * 16 + l15) * BSTRIDE];
#pragma unroll
                for (int ks = 0; ks < 4; ++ks) {
                    // B fragment: k = ks*32 + half*16 + t, t=0..15 (contiguous in LDS)
                    const __bf16* bp = bbase + ks * 32 + half * 16;
                    v8bf lo = *(const v8bf*)(bp);        // ds_load_b128
                    v8bf hi = *(const v8bf*)(bp + 8);    // ds_load_b128
                    v16bf bfrag;
#pragma unroll
                    for (int t = 0; t < 8; ++t) { bfrag[t] = lo[t]; bfrag[8 + t] = hi[t]; }
                    acc[tn] = __builtin_amdgcn_wmma_f32_16x16x32_bf16(
                        false, afr[ks], false, bfrag, (short)0, acc[tn], false, false);
                }
            }
        }
    }

    if (active) {
#pragma unroll
        for (int tn = 0; tn < tilesN; ++tn) {
            float bv = bias ? bias[tn * 16 + l15] : 0.0f;
#pragma unroll
            for (int v = 0; v < 8; ++v) {
                int row = rowTile * 16 + half * 8 + v;   // C/D layout: VGPR v -> M = v + 8*half
                C[(size_t)row * NN + tn * 16 + l15] = acc[tn][v] + bv;
            }
        }
    }
}

// ---------------------------------------------------------------------------
// el = z @ attn_l, er = z @ attn_r ; one wave per row, one float4 per lane.
// ---------------------------------------------------------------------------
__global__ void attn_dots_kernel(const float* __restrict__ z,
                                 const float* __restrict__ al,
                                 const float* __restrict__ ar,
                                 float* __restrict__ el, float* __restrict__ er,
                                 int nrows) {
    int wave = (int)((blockIdx.x * blockDim.x + threadIdx.x) >> 5);
    int lane = threadIdx.x & 31;
    if (wave >= nrows) return;
    float4 v = ((const float4*)(z + (size_t)wave * HID))[lane];
    float4 a = ((const float4*)al)[lane];
    float4 b = ((const float4*)ar)[lane];
    float sl = v.x * a.x + v.y * a.y + v.z * a.z + v.w * a.w;
    float sr = v.x * b.x + v.y * b.y + v.z * b.z + v.w * b.w;
#pragma unroll
    for (int off = 16; off > 0; off >>= 1) {
        sl += __shfl_xor(sl, off, 32);
        sr += __shfl_xor(sr, off, 32);
    }
    if (lane == 0) { el[wave] = sl; er[wave] = sr; }
}

// ---------------------------------------------------------------------------
// Init: agg=0, segsum=0, segmax=-inf, scalars {maxp,maxn,lossp,lossn}
// ---------------------------------------------------------------------------
__global__ void init_kernel(float* __restrict__ agg, float* __restrict__ ss,
                            float* __restrict__ mx, float* __restrict__ sc, int n) {
    long long i = (long long)blockIdx.x * blockDim.x + threadIdx.x;
    long long total = (long long)n * HID;
    if (i < total) agg[i] = 0.0f;
    if (i < n) { ss[i] = 0.0f; mx[i] = -INFINITY; }
    if (i == 0) { sc[0] = -INFINITY; sc[1] = -INFINITY; sc[2] = 0.0f; sc[3] = 0.0f; }
}

__device__ __forceinline__ void atomicMaxFloat(float* addr, float val) {
    // sign-aware max; valid with -inf init (bit 0xFF800000)
    if (val >= 0.0f) atomicMax((int*)addr, __float_as_int(val));
    else             atomicMin((unsigned int*)addr, __float_as_uint(val));
}

__device__ __forceinline__ float leaky02(float x) { return x > 0.0f ? x : 0.2f * x; }

// ---------------------------------------------------------------------------
// Edge pass 1: per-dst segment max of leaky_relu(el[src]+er[dst])
// ---------------------------------------------------------------------------
__global__ void edge_max_kernel(const int* __restrict__ src, const int* __restrict__ dst,
                                const float* __restrict__ el, const float* __restrict__ er,
                                float* __restrict__ mx, int E) {
    int i = (int)(blockIdx.x * blockDim.x + threadIdx.x);
    if (i >= E) return;
    float e = leaky02(el[src[i]] + er[dst[i]]);
    atomicMaxFloat(&mx[dst[i]], e);
}

// ---------------------------------------------------------------------------
// Edge pass 2: ex = exp(e - max[dst]); segment sum
// ---------------------------------------------------------------------------
__global__ void edge_exp_kernel(const int* __restrict__ src, const int* __restrict__ dst,
                                const float* __restrict__ el, const float* __restrict__ er,
                                const float* __restrict__ mx, float* __restrict__ ex,
                                float* __restrict__ ss, int E) {
    int i = (int)(blockIdx.x * blockDim.x + threadIdx.x);
    if (i >= E) return;
    int d = dst[i];
    float e = leaky02(el[src[i]] + er[d]);
    float v = __expf(e - mx[d]);
    ex[i] = v;
    atomicAdd(&ss[d], v);
}

// ---------------------------------------------------------------------------
// Edge pass 3: agg[dst] += z[src] * (ex / ssum[dst]); one wave per edge.
// z/agg are L2-resident (61 MB each on a 192 MB L2) -> scattered f32 atomics
// stay on-chip; float4 per lane covers all 128 dims with one wave.
// ---------------------------------------------------------------------------
__global__ void edge_scatter_kernel(const int* __restrict__ src, const int* __restrict__ dst,
                                    const float* __restrict__ ex, const float* __restrict__ ss,
                                    const float* __restrict__ z, float* __restrict__ agg, int E) {
    long long wave = ((long long)blockIdx.x * blockDim.x + threadIdx.x) >> 5;
    int lane = threadIdx.x & 31;
    if (wave >= E) return;
    int s = src[wave], d = dst[wave];
    float a = ex[wave] / ss[d];
    float4 v = ((const float4*)(z + (size_t)s * HID))[lane];
    float* out = agg + (size_t)d * HID + lane * 4;
    atomicAdd(out + 0, v.x * a);
    atomicAdd(out + 1, v.y * a);
    atomicAdd(out + 2, v.z * a);
    atomicAdd(out + 3, v.w * a);
}

// ---------------------------------------------------------------------------
// h = elu(agg + gat_bias), in place
// ---------------------------------------------------------------------------
__global__ void finalize_h_kernel(float* __restrict__ h, const float* __restrict__ bias, int n) {
    long long i = (long long)blockIdx.x * blockDim.x + threadIdx.x;
    if (i >= (long long)n * HID) return;
    float x = h[i] + bias[i & (HID - 1)];
    h[i] = x > 0.0f ? x : (__expf(x) - 1.0f);
}

// ---------------------------------------------------------------------------
// score = dot(h[a], h[b]); l = logsigmoid(score); track global max(l).
// One wave per pair.
// ---------------------------------------------------------------------------
__global__ void pair_score_kernel(const float* __restrict__ h,
                                  const int* __restrict__ ia, const int* __restrict__ ib,
                                  float* __restrict__ lout, float* __restrict__ gmax, int P) {
    long long wave = ((long long)blockIdx.x * blockDim.x + threadIdx.x) >> 5;
    int lane = threadIdx.x & 31;
    if (wave >= P) return;
    float4 a = ((const float4*)(h + (size_t)ia[wave] * HID))[lane];
    float4 b = ((const float4*)(h + (size_t)ib[wave] * HID))[lane];
    float s = a.x * b.x + a.y * b.y + a.z * b.z + a.w * b.w;
#pragma unroll
    for (int off = 16; off > 0; off >>= 1) s += __shfl_xor(s, off, 32);
    if (lane == 0) {
        // stable log-sigmoid: min(s,0) - log1p(exp(-|s|))
        float l = fminf(s, 0.0f) - log1pf(__expf(-fabsf(s)));
        lout[wave] = l;
        atomicMaxFloat(gmax, l);
    }
}

// ---------------------------------------------------------------------------
// hold loss: drop l > k_T*max(l), accumulate -sum(l). LDS block reduction.
// ---------------------------------------------------------------------------
__global__ void hold_loss_kernel(const float* __restrict__ l, const float* __restrict__ gmax,
                                 const int* __restrict__ alpha, float* __restrict__ acc, int P) {
    __shared__ float red[256];
    int i = (int)(blockIdx.x * blockDim.x + threadIdx.x);
    float kT = fminf(0.5f, (float)alpha[0]);
    float hold = kT * gmax[0];
    float v = 0.0f;
    if (i < P) {
        float li = l[i];
        v = (li > hold) ? 0.0f : -li;
    }
    red[threadIdx.x] = v;
    __syncthreads();
#pragma unroll
    for (int s = 128; s > 0; s >>= 1) {
        if ((int)threadIdx.x < s) red[threadIdx.x] += red[threadIdx.x + s];
        __syncthreads();
    }
    if (threadIdx.x == 0) atomicAdd(acc, red[0]);
}

__global__ void combine_loss_kernel(const float* __restrict__ sc, float* __restrict__ out) {
    if (threadIdx.x == 0 && blockIdx.x == 0) out[0] = sc[2] + sc[3];
}

// ---------------------------------------------------------------------------

static inline int cdiv(long long a, long long b) { return (int)((a + b - 1) / b); }

extern "C" void kernel_launch(void* const* d_in, const int* in_sizes, int n_in,
                              void* d_out, int out_size, void* d_ws, size_t ws_size,
                              hipStream_t stream) {
    const float* feat_user = (const float*)d_in[0];
    const float* feat_item = (const float*)d_in[1];
    const float* W_user    = (const float*)d_in[2];
    const float* b_user    = (const float*)d_in[3];
    const float* W_item    = (const float*)d_in[4];
    const float* b_item    = (const float*)d_in[5];
    const float* W_gat     = (const float*)d_in[6];
    const float* attn_l    = (const float*)d_in[7];
    const float* attn_r    = (const float*)d_in[8];
    const float* gat_bias  = (const float*)d_in[9];
    const float* W_pred    = (const float*)d_in[10];
    const float* b_pred    = (const float*)d_in[11];
    const int*   src       = (const int*)d_in[12];
    const int*   dst       = (const int*)d_in[13];
    const int*   pos_src   = (const int*)d_in[14];
    const int*   pos_dst   = (const int*)d_in[15];
    const int*   neg_src   = (const int*)d_in[16];
    const int*   neg_dst   = (const int*)d_in[17];
    const int*   alpha     = (const int*)d_in[18];

    const int Nu = in_sizes[0] / 256;   // 60000
    const int Ni = in_sizes[1] / 128;   // 60000
    const int N  = Nu + Ni;             // 120000
    const int E  = in_sizes[12];        // 1,620,000
    const int P  = in_sizes[14];        // 200,000

    // workspace layout (floats, then bf16 transposed weights)
    float* ws = (float*)d_ws;
    size_t o = 0;
    float* x   = ws + o; o += (size_t)N * HID;   // also reused as agg/h after z
    float* z   = ws + o; o += (size_t)N * HID;
    float* el  = ws + o; o += (size_t)N;
    float* er  = ws + o; o += (size_t)N;
    float* mx  = ws + o; o += (size_t)N;
    float* ss  = ws + o; o += (size_t)N;
    float* ex  = ws + o; o += (size_t)E;
    float* lp  = ws + o; o += (size_t)P;
    float* ln  = ws + o; o += (size_t)P;
    float* sc  = ws + o; o += 8;                 // maxp, maxn, lossp, lossn
    o = (o + 7) & ~(size_t)7;                    // 16B align for bf16 region
    __bf16* wt_user = (__bf16*)(ws + o);         // [128][256]
    __bf16* wt_item = wt_user + 128 * 256;       // [128][128]
    __bf16* wt_gat  = wt_item + 128 * 128;       // [128][128]
    __bf16* wt_pred = wt_gat  + 128 * 128;       // [64][128]

    float* dout = (float*)d_out;                 // [0]=loss, [1:]=pred out (60000x64)

    const int TB = 256;
    // --- one-time weight transpose+convert (tiny) ---
    wt_transpose_kernel<<<cdiv(256 * 128, TB), TB, 0, stream>>>(W_user, wt_user, 256, 128);
    wt_transpose_kernel<<<cdiv(128 * 128, TB), TB, 0, stream>>>(W_item, wt_item, 128, 128);
    wt_transpose_kernel<<<cdiv(128 * 128, TB), TB, 0, stream>>>(W_gat,  wt_gat,  128, 128);
    wt_transpose_kernel<<<cdiv(128 * 64,  TB), TB, 0, stream>>>(W_pred, wt_pred, 128, 64);

    // Each GEMM block covers 128 M-rows (8 wave-tiles).
    // --- input projections (WMMA) ---
    wmma_gemm_kernel<256, 128><<<cdiv(cdiv(Nu, 16), 8), TB, 0, stream>>>(
        feat_user, wt_user, b_user, x, Nu);
    wmma_gemm_kernel<128, 128><<<cdiv(cdiv(Ni, 16), 8), TB, 0, stream>>>(
        feat_item, wt_item, b_item, x + (size_t)Nu * HID, Ni);
    // --- z = x @ W_gat (no bias) ---
    wmma_gemm_kernel<128, 128><<<cdiv(cdiv(N, 16), 8), TB, 0, stream>>>(
        x, wt_gat, nullptr, z, N);
    // --- el / er ---
    attn_dots_kernel<<<cdiv((long long)N * 32, TB), TB, 0, stream>>>(z, attn_l, attn_r, el, er, N);
    // --- init accumulators (x becomes agg from here) ---
    init_kernel<<<cdiv((long long)N * HID, TB), TB, 0, stream>>>(x, ss, mx, sc, N);
    // --- edge softmax + scatter ---
    edge_max_kernel<<<cdiv(E, TB), TB, 0, stream>>>(src, dst, el, er, mx, E);
    edge_exp_kernel<<<cdiv(E, TB), TB, 0, stream>>>(src, dst, el, er, mx, ex, ss, E);
    edge_scatter_kernel<<<cdiv((long long)E * 32, TB), TB, 0, stream>>>(src, dst, ex, ss, z, x, E);
    // --- h = elu(agg + bias) ---
    finalize_h_kernel<<<cdiv((long long)N * HID, TB), TB, 0, stream>>>(x, gat_bias, N);
    // --- prediction head (WMMA): out = h[:Nu] @ W_pred + b_pred ---
    wmma_gemm_kernel<128, 64><<<cdiv(cdiv(Nu, 16), 8), TB, 0, stream>>>(
        x, wt_pred, b_pred, dout + 1, Nu);
    // --- link scores + hold loss ---
    pair_score_kernel<<<cdiv((long long)P * 32, TB), TB, 0, stream>>>(x, pos_src, pos_dst, lp, sc + 0, P);
    pair_score_kernel<<<cdiv((long long)P * 32, TB), TB, 0, stream>>>(x, neg_src, neg_dst, ln, sc + 1, P);
    hold_loss_kernel<<<cdiv(P, TB), TB, 0, stream>>>(lp, sc + 0, alpha, sc + 2, P);
    hold_loss_kernel<<<cdiv(P, TB), TB, 0, stream>>>(ln, sc + 1, alpha, sc + 3, P);
    combine_loss_kernel<<<1, 1, 0, stream>>>(sc, dout);

    (void)n_in; (void)out_size; (void)ws_size;
}
